// AliBiCausalSelfAttentionWithCache_49426483642590
// MI455X (gfx1250) — matrix operand order repaired and checked by reading, not compiled, not run on hardware
//
#include <hip/hip_runtime.h>
#include <hip/hip_bf16.h>

// ---------------------------------------------------------------------------
// CDNA5 (gfx1250, wave32) WMMA 16x16x32 bf16 -> f32
// ---------------------------------------------------------------------------
typedef __attribute__((ext_vector_type(16))) __bf16        v16bf;
typedef __attribute__((ext_vector_type(8)))  float         v8f;
typedef __attribute__((ext_vector_type(8)))  unsigned int  v8u;

#ifndef USE_ASYNC_LDS
#define USE_ASYNC_LDS 1   // global_load_async_to_lds_b128 staging (ASYNCcnt path)
#endif

__device__ __forceinline__ unsigned int bf16_1(float f) {
    unsigned int u = __builtin_bit_cast(unsigned int, f);
    return (u + 0x7FFFu + ((u >> 16) & 1u)) >> 16;   // RNE to bf16
}
__device__ __forceinline__ unsigned int bf16_pack(float lo, float hi) {
    return (bf16_1(lo) & 0xFFFFu) | (bf16_1(hi) << 16);
}
__device__ __forceinline__ v8f wmma_bf16(v8u a, v8u b, v8f c) {
    return __builtin_amdgcn_wmma_f32_16x16x32_bf16(
        false, __builtin_bit_cast(v16bf, a),
        false, __builtin_bit_cast(v16bf, b),
        (short)0, c, false, false);
}

// 16-byte global->LDS copy: async DMA on CDNA5 (bypasses VGPRs, ASYNCcnt).
__device__ __forceinline__ void cp16_g2l(const unsigned int* gsrc, unsigned int* ldst) {
#if USE_ASYNC_LDS
    const __attribute__((address_space(1))) unsigned int* g =
        (const __attribute__((address_space(1))) unsigned int*)gsrc;
    __attribute__((address_space(3))) unsigned int* l =
        (__attribute__((address_space(3))) unsigned int*)ldst;
    asm volatile("global_load_async_to_lds_b128 %0, %1, off"
                 :: "v"(l), "v"(g) : "memory");
#else
    *(uint4*)ldst = *(const uint4*)gsrc;
#endif
}
__device__ __forceinline__ void wait_stage_prev() {   // retire previous tile (4 ops)
#if USE_ASYNC_LDS
    asm volatile("s_wait_asynccnt 0x4" ::: "memory");
#endif
}
__device__ __forceinline__ void wait_stage_all() {
#if USE_ASYNC_LDS
    asm volatile("s_wait_asynccnt 0x0" ::: "memory");
#endif
}

// Row stride in uints: 80B rows -> 16B-aligned for b128 async stores, and
// 20*r mod 64 = 4*(5r mod 16) is a permutation -> conflict-free frag reads.
#define LDS_STRIDE 20
#define P_STRIDE   17   // attention P tile (b32 accesses only)

// ---------------------------------------------------------------------------
// Pre-conversion kernels (run once; GEMMs then move only bf16 bytes).
// ---------------------------------------------------------------------------
__global__ __launch_bounds__(256) void convert_pairs_bf16(
    const float* __restrict__ src, unsigned int* __restrict__ dst, int npairs)
{
    int i = blockIdx.x * 256 + threadIdx.x;
    if (i < npairs) dst[i] = bf16_pack(src[2 * i], src[2 * i + 1]);
}

// W[K][N] f32 -> WT[N][K] bf16 via 64x64 LDS tile (coalesced both sides).
__global__ __launch_bounds__(256) void transpose_convert_bf16(
    const float* __restrict__ W, unsigned int* __restrict__ WTu, int K, int N)
{
    __shared__ unsigned short t[64][65];
    const int tid = threadIdx.x;
    const int n0 = blockIdx.x * 64, k0 = blockIdx.y * 64;
#pragma unroll
    for (int e = 0; e < 16; ++e) {
        int idx = tid + 256 * e;
        int i = idx >> 6, j = idx & 63;            // i: k, j: n (coalesced on j)
        t[j][i] = (unsigned short)bf16_1(W[(size_t)(k0 + i) * N + (n0 + j)]);
    }
    __syncthreads();
    const int Ku = K >> 1;
#pragma unroll
    for (int e = 0; e < 8; ++e) {
        int idx = tid + 256 * e;                   // 2048 uint stores
        int j = idx >> 5, ic = idx & 31;           // j: n, ic: k-pair (coalesced)
        unsigned int v = (unsigned int)t[j][2 * ic] | ((unsigned int)t[j][2 * ic + 1] << 16);
        WTu[(size_t)(n0 + j) * Ku + (k0 >> 1) + ic] = v;
    }
}

// ---------------------------------------------------------------------------
// Tiled bf16 GEMM: C[M,N] = A[M,K] * WT[N,K]^T + bias.
// 256 threads (8 waves), 128x128 tile, wave = 32x64, double-buffered LDS
// staged with async global->LDS b128 copies.
// mode 0: f32 C -> outF.   mode 1: QKV split epilogue.
// ---------------------------------------------------------------------------
__global__ __launch_bounds__(256) void gemm_bf16_wmma(
    const unsigned short* __restrict__ Abf,   // [M][K] bf16
    const unsigned short* __restrict__ WT,    // [N][K] bf16
    const float* __restrict__ bias, int Kdim, int N, int mode,
    float* __restrict__ outF,
    unsigned short* __restrict__ qws, unsigned short* __restrict__ kws,
    unsigned short* __restrict__ vTs,
    float* __restrict__ koutF, float* __restrict__ voutF)
{
    __shared__ unsigned int lA[2][128 * LDS_STRIDE];
    __shared__ unsigned int lB[2][128 * LDS_STRIDE];

    const int tid  = threadIdx.x;
    const int lane = tid & 31;
    const int wave = tid >> 5;
    const int wm   = wave >> 1, wn = wave & 1;
    const int bm   = blockIdx.y * 128, bn = blockIdx.x * 128;
    const int l15  = lane & 15;
    const int g4   = (lane >= 16) ? 4 : 0;
    const int g8   = (lane >= 16) ? 8 : 0;

    const unsigned int* Au = (const unsigned int*)Abf;
    const unsigned int* Wu = (const unsigned int*)WT;
    const int Ku = Kdim >> 1;                 // uints per row
    const int NT = Kdim >> 5;                 // 32-wide k tiles

    // stage tile kt into buffer buf: 4 async b128 ops per thread total
    auto stage = [&](int kt, int buf) {
        const int k0u = kt << 4;
#pragma unroll
        for (int e = 0; e < 2; ++e) {
            int idx = tid + 256 * e;
            int row = idx >> 2, ch = (idx & 3) << 2;
            cp16_g2l(Au + (size_t)(bm + row) * Ku + k0u + ch,
                     &lA[buf][row * LDS_STRIDE + ch]);
        }
#pragma unroll
        for (int e = 0; e < 2; ++e) {
            int idx = tid + 256 * e;
            int row = idx >> 2, ch = (idx & 3) << 2;
            cp16_g2l(Wu + (size_t)(bn + row) * Ku + k0u + ch,
                     &lB[buf][row * LDS_STRIDE + ch]);
        }
    };

    v8f acc[2][4];
#pragma unroll
    for (int i = 0; i < 2; ++i)
#pragma unroll
        for (int j = 0; j < 4; ++j)
#pragma unroll
            for (int r = 0; r < 8; ++r) acc[i][j][r] = 0.0f;

    stage(0, 0);
    for (int kt = 0; kt < NT; ++kt) {
        const int buf = kt & 1;
        if (kt + 1 < NT) { stage(kt + 1, buf ^ 1); wait_stage_prev(); }
        else             { wait_stage_all(); }
        __syncthreads();

        v8u af[2], bfr[4];
#pragma unroll
        for (int ms = 0; ms < 2; ++ms) {
            const unsigned int* r = &lA[buf][(wm * 32 + ms * 16 + l15) * LDS_STRIDE];
#pragma unroll
            for (int v = 0; v < 8; ++v)
                af[ms][v] = r[(v < 4) ? (g4 + v) : (8 + g4 + (v - 4))];
        }
#pragma unroll
        for (int ns = 0; ns < 4; ++ns) {
            const unsigned int* r = &lB[buf][(wn * 64 + ns * 16 + l15) * LDS_STRIDE];
#pragma unroll
            for (int v = 0; v < 8; ++v)
                bfr[ns][v] = r[(v < 4) ? (g4 + v) : (8 + g4 + (v - 4))];
        }
#pragma unroll
        for (int ms = 0; ms < 2; ++ms)
#pragma unroll
            for (int ns = 0; ns < 4; ++ns)
                acc[ms][ns] = wmma_bf16(af[ms], bfr[ns], acc[ms][ns]);
        __syncthreads();
    }

    // epilogue: C layout = VGPR r -> row (r + g8), col = l15
#pragma unroll
    for (int ms = 0; ms < 2; ++ms) {
#pragma unroll
        for (int ns = 0; ns < 4; ++ns) {
            int gn = bn + wn * 64 + ns * 16 + l15;
            float bv = bias[gn];
#pragma unroll
            for (int r = 0; r < 8; ++r) {
                int gm = bm + wm * 32 + ms * 16 + r + g8;
                float val = acc[ms][ns][r] + bv;
                if (mode == 0) {
                    outF[(size_t)gm * N + gn] = val;
                } else {
                    int region = gn >> 10;        // 0=q, 1=k, 2=v
                    int dcol = gn & 1023;
                    int hh = dcol >> 6, dh = dcol & 63;
                    int bb = gm >> 11, ll = gm & 2047;
                    int bh = (bb << 4) | hh;
                    size_t lin = ((size_t)((bh << 11) | ll) << 6) | dh;
                    unsigned short v16 = (unsigned short)bf16_1(val);
                    if (region == 0)      { qws[lin] = v16; }
                    else if (region == 1) { kws[lin] = v16; koutF[lin] = val; }
                    else {
                        vTs[(((size_t)((bh << 6) | dh)) << 11) | ll] = v16;
                        voutF[lin] = val;
                    }
                }
            }
        }
    }
}

// ---------------------------------------------------------------------------
// Flash attention with ALiBi + causal mask. 1024 blocks x 4 waves; per
// 32-key tile: 4 WMMAs (Q K^T) + online softmax + 4 WMMAs (P V).
// ---------------------------------------------------------------------------
__global__ __launch_bounds__(128) void attn_flash(
    const unsigned short* __restrict__ qbh,   // [BH, L, 64] bf16
    const unsigned short* __restrict__ kbh,   // [BH, L, 64] bf16
    const unsigned short* __restrict__ vT,    // [BH, 64, L] bf16
    unsigned short* __restrict__ aout)        // [B, L, 1024] bf16
{
    __shared__ unsigned int lP[4][16 * P_STRIDE];

    const int tid  = threadIdx.x;
    const int lane = tid & 31;
    const int wave = tid >> 5;
    const int l15  = lane & 15;
    const int g4   = (lane >= 16) ? 4 : 0;
    const int g8   = (lane >= 16) ? 8 : 0;

    const int qblk = blockIdx.x & 31;
    const int bh   = blockIdx.x >> 5;
    const int h    = bh & 15;
    const int b    = bh >> 4;
    const int q0   = qblk * 64;
    const int qw   = q0 + wave * 16;
    const float slope = exp2f(-0.5f * (float)(h + 1));
    const float scale = 0.125f;

    v8u Aq[2];
    {
        const unsigned int* qrow =
            (const unsigned int*)(qbh + ((size_t)((bh << 11) | (qw + l15)) << 6));
#pragma unroll
        for (int kk = 0; kk < 2; ++kk)
#pragma unroll
            for (int v = 0; v < 8; ++v)
                Aq[kk][v] = qrow[kk * 16 + ((v < 4) ? (g4 + v) : (8 + g4 + (v - 4)))];
    }

    v8f O[4];
    float m_i[8], l_i[8];
#pragma unroll
    for (int c = 0; c < 4; ++c)
#pragma unroll
        for (int r = 0; r < 8; ++r) O[c][r] = 0.0f;
#pragma unroll
    for (int r = 0; r < 8; ++r) { m_i[r] = -1e30f; l_i[r] = 0.0f; }

    const int ntiles = (q0 + 64) >> 5;   // uniform per block
    for (int j = 0; j < ntiles; ++j) {
        const int key0 = j << 5;

        v8f S[2];
#pragma unroll
        for (int nh = 0; nh < 2; ++nh) {
#pragma unroll
            for (int r = 0; r < 8; ++r) S[nh][r] = 0.0f;
            const unsigned int* krow = (const unsigned int*)
                (kbh + ((size_t)((bh << 11) | (key0 + nh * 16 + l15)) << 6));
            v8u B0, B1;
#pragma unroll
            for (int v = 0; v < 8; ++v) { int o = (v < 4) ? (g4 + v) : (8 + g4 + (v - 4)); B0[v] = krow[o]; B1[v] = krow[16 + o]; }
            S[nh] = wmma_bf16(Aq[0], B0, S[nh]);
            S[nh] = wmma_bf16(Aq[1], B1, S[nh]);
        }

        unsigned int* Pu = &lP[wave][0];
        unsigned short* Ps = (unsigned short*)Pu;
#pragma unroll
        for (int r = 0; r < 8; ++r) {
            int row = qw + r + g8;
            int ka = key0 + l15, kb = key0 + 16 + l15;
            float s0 = S[0][r] * scale + slope * (float)(ka - row);
            float s1 = S[1][r] * scale + slope * (float)(kb - row);
            if (ka > row) s0 = -1e30f;
            if (kb > row) s1 = -1e30f;
            float mx = fmaxf(s0, s1);
            mx = fmaxf(mx, __shfl_xor(mx, 1, 16));
            mx = fmaxf(mx, __shfl_xor(mx, 2, 16));
            mx = fmaxf(mx, __shfl_xor(mx, 4, 16));
            mx = fmaxf(mx, __shfl_xor(mx, 8, 16));
            float mnew  = fmaxf(m_i[r], mx);
            float p0    = __expf(s0 - mnew);
            float p1    = __expf(s1 - mnew);
            float alpha = __expf(m_i[r] - mnew);
            m_i[r] = mnew;
            float rs = p0 + p1;
            rs += __shfl_xor(rs, 1, 16);
            rs += __shfl_xor(rs, 2, 16);
            rs += __shfl_xor(rs, 4, 16);
            rs += __shfl_xor(rs, 8, 16);
            l_i[r] = l_i[r] * alpha + rs;
#pragma unroll
            for (int c = 0; c < 4; ++c) O[c][r] *= alpha;
            Ps[(r + g8) * (2 * P_STRIDE) + l15]      = (unsigned short)bf16_1(p0);
            Ps[(r + g8) * (2 * P_STRIDE) + 16 + l15] = (unsigned short)bf16_1(p1);
        }
        __syncthreads();

        v8u Ap;
        const unsigned int* prow = &Pu[l15 * P_STRIDE];
#pragma unroll
        for (int v = 0; v < 8; ++v)
            Ap[v] = prow[(v < 4) ? (g4 + v) : (8 + g4 + (v - 4))];

#pragma unroll
        for (int c = 0; c < 4; ++c) {
            const unsigned int* vrow = (const unsigned int*)
                (vT + ((size_t)((bh << 6) | (c * 16 + l15)) << 11));
            v8u Bv;
#pragma unroll
            for (int v = 0; v < 8; ++v) Bv[v] = vrow[(key0 >> 1) + ((v < 4) ? (g4 + v) : (8 + g4 + (v - 4)))];
            O[c] = wmma_bf16(Ap, Bv, O[c]);
        }
        __syncthreads();
    }

#pragma unroll
    for (int r = 0; r < 8; ++r) {
        int row = qw + r + g8;
        float inv = 1.0f / l_i[r];
#pragma unroll
        for (int c = 0; c < 4; ++c)
            aout[((size_t)((b << 11) | row) << 10) | (h * 64 + c * 16 + l15)]
                = (unsigned short)bf16_1(O[c][r] * inv);
    }
}

// ---------------------------------------------------------------------------
// Host launch: convert -> qkv GEMM -> attention -> out-proj GEMM
// ---------------------------------------------------------------------------
extern "C" void kernel_launch(void* const* d_in, const int* in_sizes, int n_in,
                              void* d_out, int out_size, void* d_ws, size_t ws_size,
                              hipStream_t stream) {
    const float* x     = (const float*)d_in[0];   // [2,2048,1024]
    const float* W_in  = (const float*)d_in[1];   // [1024,3072]
    const float* b_in  = (const float*)d_in[2];   // [3072]
    const float* W_out = (const float*)d_in[3];   // [1024,1024]
    const float* b_out = (const float*)d_in[4];   // [1024]

    const size_t BLD = 4194304;                   // B*L*D
    float* out  = (float*)d_out;
    float* kout = out + BLD;
    float* vout = out + 2 * BLD;

    unsigned short* xbf   = (unsigned short*)d_ws;     // [4096][1024] bf16
    unsigned short* wtin  = xbf   + BLD;               // [3072][1024] bf16
    unsigned short* wtout = wtin  + 3145728;           // [1024][1024] bf16
    unsigned short* qbh   = wtout + 1048576;           // [BH,L,64] bf16
    unsigned short* kbh   = qbh   + BLD;
    unsigned short* vT    = kbh   + BLD;               // [BH,64,L] bf16
    unsigned short* abf   = vT    + BLD;               // [B,L,1024] bf16

    // one-time bf16 conversion / transposition
    convert_pairs_bf16<<<dim3(8192), dim3(256), 0, stream>>>(
        x, (unsigned int*)xbf, 2097152);
    transpose_convert_bf16<<<dim3(48, 16), dim3(256), 0, stream>>>(
        W_in, (unsigned int*)wtin, 1024, 3072);
    transpose_convert_bf16<<<dim3(16, 16), dim3(256), 0, stream>>>(
        W_out, (unsigned int*)wtout, 1024, 1024);

    // 1) QKV projection [4096,1024] x [1024,3072]
    gemm_bf16_wmma<<<dim3(24, 32), dim3(256), 0, stream>>>(
        xbf, wtin, b_in, 1024, 3072, 1,
        nullptr, qbh, kbh, vT, kout, vout);

    // 2) flash attention with ALiBi
    attn_flash<<<dim3(1024), dim3(128), 0, stream>>>(qbh, kbh, vT, abf);

    // 3) output projection [4096,1024] x [1024,1024]
    gemm_bf16_wmma<<<dim3(8, 32), dim3(256), 0, stream>>>(
        abf, wtout, b_out, 1024, 1024, 0,
        out, nullptr, nullptr, nullptr, nullptr, nullptr);
}
